// NodeEncoder_15771119911318
// MI455X (gfx1250) — matrix operand order repaired and checked by reading, not compiled, not run on hardware
//
#include <hip/hip_runtime.h>

typedef __attribute__((ext_vector_type(16))) _Float16 v16h;
typedef __attribute__((ext_vector_type(8)))  float    v8f;

#define KDIM 128   // K for both layers (IN == HID == 128)

// ---------------------------------------------------------------- degree
__global__ void deg_init_kernel(int* __restrict__ deg, int n) {
    int v = blockIdx.x * blockDim.x + threadIdx.x;
    if (v < n) deg[v] = 1;               // self-loop
}

__global__ void deg_count_kernel(const int* __restrict__ dst, int* __restrict__ deg, int e) {
    int i = blockIdx.x * blockDim.x + threadIdx.x;
    if (i < e) atomicAdd(&deg[dst[i]], 1);
}

__global__ void dinv_kernel(const int* __restrict__ deg, float* __restrict__ dinv, int n) {
    int v = blockIdx.x * blockDim.x + threadIdx.x;
    if (v < n) dinv[v] = rsqrtf((float)deg[v]);   // deg >= 1 always
}

// ------------------------------------------- W [k][n] f32 -> Wt [n][k] f16
__global__ void wtrans_kernel(const float* __restrict__ W, _Float16* __restrict__ Wt,
                              int K, int Ncol) {
    int i = blockIdx.x * blockDim.x + threadIdx.x;
    if (i >= K * Ncol) return;
    int k = i / Ncol, nc = i % Ncol;
    Wt[(size_t)nc * K + k] = (_Float16)W[i];
}

// ---------------------------------------------------------------- WMMA GEMM
// One wave computes a 16 x NOUT tile of  g = (X @ W) * dinv[row],
// and also initializes acc = g (the self-loop contribution).
// B (all of Wt: 32KB/16KB f16) is staged once per workgroup into LDS so the
// inner loop reads fragments via ds_load_b128; VMEM streams only the A rows.
// A fragment (16x32 f16): lane L holds row m=L%16;
//   K = (L<16?0:16)+i+32*ki
// B fragment (32x16 f16): lane L holds col n=L%16, K = (L<16?0:16)+i+32*ki
// C/D (16x16 f32): VGPR r -> row r + (L>=16 ? 8 : 0), col L%16
template<int NOUT>
__global__ __launch_bounds__(128)
void gemm_scale_kernel(const float* __restrict__ X,        // [n, KDIM] f32
                       const _Float16* __restrict__ Wt,    // [NOUT, KDIM] f16
                       const float* __restrict__ dinv,     // [n]
                       float* __restrict__ g,              // [n, NOUT]
                       float* __restrict__ acc,            // [n, NOUT]
                       int n_nodes)
{
    __shared__ _Float16 sW[NOUT * KDIM];

    // cooperative stage of Wt into LDS (b128 chunks), before any exit
    {
        const uint4* srcW = (const uint4*)Wt;
        uint4*       dstW = (uint4*)sW;
        constexpr int chunks = NOUT * KDIM / 8;        // 8 halves per uint4
#pragma unroll 4
        for (int i = threadIdx.x; i < chunks; i += 128) dstW[i] = srcW[i];
    }
    __syncthreads();

    const int lane = threadIdx.x & 31;
    const int wave = threadIdx.x >> 5;
    const int tile = blockIdx.x * (blockDim.x >> 5) + wave;
    const int row0 = tile * 16;
    if (row0 >= n_nodes) return;          // wave-uniform exit (EXEC stays all-1)

    constexpr int NT = NOUT / 16;
    v8f zero = {};
    v8f accv[NT];
#pragma unroll
    for (int t = 0; t < NT; ++t) accv[t] = zero;

    const int  m  = lane & 15;
    const bool hi = lane >= 16;
    int arow = row0 + m;
    if (arow >= n_nodes) arow = n_nodes - 1;      // clamp, keep EXEC full
    const float* xrow = X + (size_t)arow * KDIM;

#pragma unroll
    for (int ki = 0; ki < KDIM / 32; ++ki) {
        const int klo = ki * 32 + (hi ? 16 : 0);
        const int khi = klo + 8;
        float4 c0 = *(const float4*)(xrow + klo);
        float4 c1 = *(const float4*)(xrow + klo + 4);
        float4 c2 = *(const float4*)(xrow + khi);
        float4 c3 = *(const float4*)(xrow + khi + 4);
        v16h a;
        a[0]=(_Float16)c0.x; a[1]=(_Float16)c0.y; a[2]=(_Float16)c0.z; a[3]=(_Float16)c0.w;
        a[4]=(_Float16)c1.x; a[5]=(_Float16)c1.y; a[6]=(_Float16)c1.z; a[7]=(_Float16)c1.w;
        a[8]=(_Float16)c2.x; a[9]=(_Float16)c2.y; a[10]=(_Float16)c2.z; a[11]=(_Float16)c2.w;
        a[12]=(_Float16)c3.x; a[13]=(_Float16)c3.y; a[14]=(_Float16)c3.z; a[15]=(_Float16)c3.w;

        const int koffB = ki * 32 + (hi ? 16 : 0);
#pragma unroll
        for (int t = 0; t < NT; ++t) {
            // contiguous 32B LDS read (2x ds_load_b128)
            v16h b = *(const v16h*)(sW + (t * 16 + m) * KDIM + koffB);
            accv[t] = __builtin_amdgcn_wmma_f32_16x16x32_f16(
                false, a, false, b, (short)0, accv[t], false, false);
        }
    }

    // epilogue: scale by dinv, write g and acc (= self-loop init)
#pragma unroll
    for (int r = 0; r < 8; ++r) {
        const int mr  = r + (hi ? 8 : 0);
        const int row = row0 + mr;
        if (row < n_nodes) {
            const float dv = dinv[row];
#pragma unroll
            for (int t = 0; t < NT; ++t) {
                float val = accv[t][r] * dv;
                size_t off = (size_t)row * NOUT + t * 16 + m;
                g[off]   = val;
                acc[off] = val;
            }
        }
    }
}

// ---------------------------------------------------------------- scatter
// One wave per edge: coalesced 512B/256B row gather + f32 atomic scatter-add.
// g/acc are L2-resident (51MB << 192MB L2), so this is L2-bandwidth bound.
template<int D>
__global__ void scatter_kernel(const int* __restrict__ src,
                               const int* __restrict__ dst,
                               const float* __restrict__ g,
                               float* __restrict__ acc,
                               int n_edges)
{
    const int lane = threadIdx.x & 31;
    const int e = (int)((blockIdx.x * (size_t)blockDim.x + threadIdx.x) >> 5);
    if (e >= n_edges) return;
    const int s = src[e];
    const int d = dst[e];
    constexpr int PER = D / 32;                    // 4 (D=128) or 2 (D=64)
    const float* gp = g   + (size_t)s * D + lane * PER;
    float*       ap = acc + (size_t)d * D + lane * PER;
    float v[PER];
#pragma unroll
    for (int i = 0; i < PER; ++i) v[i] = gp[i];    // wide L2-resident gather
#pragma unroll
    for (int i = 0; i < PER; ++i) atomicAdd(ap + i, v[i]);  // non-returning
}

// ---------------------------------------------------------------- finalize
__global__ void finalize_kernel(const float* __restrict__ acc,
                                const float* __restrict__ dinv,
                                const float* __restrict__ bias,
                                float* __restrict__ out,
                                int n_nodes, int d, int relu)
{
    size_t i = blockIdx.x * (size_t)blockDim.x + threadIdx.x;
    if (i >= (size_t)n_nodes * d) return;
    int v = (int)(i / d);
    int c = (int)(i % d);
    float val = acc[i] * dinv[v] + bias[c];
    if (relu) val = fmaxf(val, 0.0f);
    out[i] = val;
}

// ---------------------------------------------------------------- launch
extern "C" void kernel_launch(void* const* d_in, const int* in_sizes, int n_in,
                              void* d_out, int out_size, void* d_ws, size_t ws_size,
                              hipStream_t stream) {
    const float* x    = (const float*)d_in[0];   // [n,128]
    const float* W1   = (const float*)d_in[1];   // [128,128]
    const float* b1   = (const float*)d_in[2];   // [128]
    const float* W2   = (const float*)d_in[3];   // [128,64]
    const float* b2   = (const float*)d_in[4];   // [64]
    const int*   edge = (const int*)d_in[5];     // [2,E]

    const int n = in_sizes[0] / 128;
    const int e = in_sizes[5] / 2;
    const int* srcp = edge;
    const int* dstp = edge + e;
    float* out = (float*)d_out;

    // carve workspace (all re-initialized every call; ~103 MB total)
    char* ws = (char*)d_ws;
    auto carve = [&](size_t bytes) {
        char* p = ws;
        ws += (bytes + 255) & ~(size_t)255;
        return p;
    };
    int*       deg  = (int*)      carve((size_t)n * 4);
    float*     dinv = (float*)    carve((size_t)n * 4);
    _Float16*  W1t  = (_Float16*) carve((size_t)128 * 128 * 2);
    _Float16*  W2t  = (_Float16*) carve((size_t)64  * 128 * 2);
    float*     g1   = (float*)    carve((size_t)n * 128 * 4);  // later reused as h1
    float*     acc1 = (float*)    carve((size_t)n * 128 * 4);  // later reused as g2+acc2
    float*     h1   = g1;
    float*     g2   = acc1;
    float*     acc2 = acc1 + (size_t)n * 64;

    const int T = 256;
    // degrees + normalization
    deg_init_kernel <<<(n + T - 1) / T, T, 0, stream>>>(deg, n);
    deg_count_kernel<<<(e + T - 1) / T, T, 0, stream>>>(dstp, deg, e);
    dinv_kernel     <<<(n + T - 1) / T, T, 0, stream>>>(deg, dinv, n);

    // weight transpose+convert to f16 [n][k]
    wtrans_kernel<<<(128 * 128 + T - 1) / T, T, 0, stream>>>(W1, W1t, 128, 128);
    wtrans_kernel<<<(128 * 64  + T - 1) / T, T, 0, stream>>>(W2, W2t, 128, 64);

    const int tiles  = (n + 15) / 16;
    const int gblk   = (tiles + 3) / 4;          // 4 waves (128 threads) per block
    const size_t ethreads = (size_t)e * 32;

    // ---- layer 1 ----
    gemm_scale_kernel<128><<<gblk, 128, 0, stream>>>(x, W1t, dinv, g1, acc1, n);
    scatter_kernel<128><<<(unsigned)((ethreads + T - 1) / T), T, 0, stream>>>(srcp, dstp, g1, acc1, e);
    finalize_kernel<<<(unsigned)(((size_t)n * 128 + T - 1) / T), T, 0, stream>>>(
        acc1, dinv, b1, h1, n, 128, 1);

    // ---- layer 2 ----
    gemm_scale_kernel<64><<<gblk, 128, 0, stream>>>(h1, W2t, dinv, g2, acc2, n);
    scatter_kernel<64><<<(unsigned)((ethreads + T - 1) / T), T, 0, stream>>>(srcp, dstp, g2, acc2, e);
    finalize_kernel<<<(unsigned)(((size_t)n * 64 + T - 1) / T), T, 0, stream>>>(
        acc2, dinv, b2, out, n, 64, 0);
}
